// _SpatialAttention2D_45449343926365
// MI455X (gfx1250) — compile-verified, matrix operand order checked by reading, not run on hardware
//
#include <hip/hip_runtime.h>

// ---------------------------------------------------------------------------
// SpatialAttention2D for MI455X (gfx1250, wave32, WMMA bf16 16x16x32)
// B=8, C=256, H=W=64 -> n=4096, HID=32
// ---------------------------------------------------------------------------

typedef __attribute__((ext_vector_type(16))) __bf16 v16bf;
typedef __attribute__((ext_vector_type(8)))  __bf16 v8bf;
typedef __attribute__((ext_vector_type(8)))  float  v8f;
typedef __attribute__((ext_vector_type(4)))  int    v4i;

#define BATCH 8
#define CH    256
#define NPIX  4096
#define HID   32

// ---- async global->LDS copy (CDNA5), guarded for toolchain portability ----
#if defined(__has_builtin)
#if __has_builtin(__builtin_amdgcn_global_load_async_to_lds_b128)
#define HAVE_ASYNC_LDS 1
#endif
#endif

static __device__ __forceinline__ void copy16_to_lds(void* lds_dst, const void* gsrc) {
#ifdef HAVE_ASYNC_LDS
  __builtin_amdgcn_global_load_async_to_lds_b128(
      (__attribute__((address_space(1))) v4i*)(v4i*)const_cast<void*>(gsrc),
      (__attribute__((address_space(3))) v4i*)(v4i*)lds_dst,
      0, 0);
#else
  *(uint4*)lds_dst = *(const uint4*)gsrc;
#endif
}

static __device__ __forceinline__ void async_copies_wait() {
#ifdef HAVE_ASYNC_LDS
#if __has_builtin(__builtin_amdgcn_s_wait_asynccnt)
  __builtin_amdgcn_s_wait_asynccnt(0);
#else
  asm volatile("s_wait_asynccnt 0" ::: "memory");
#endif
#endif
}

static __device__ __forceinline__ v16bf cat16(v8bf lo, v8bf hi) {
  v16bf r;
#pragma unroll
  for (int e = 0; e < 8; ++e) { r[e] = lo[e]; r[8 + e] = hi[e]; }
  return r;
}

// ---------------------------------------------------------------------------
// Projection GEMM: Y[m][n] = sum_k W[m][k] * X[b][k][n] + bias[m]
// One wave computes one 16x16 output tile; K=256 in 8 WMMA steps of 32.
// STORE_T=true  -> Y stored [b][n][M] (pixel-major, for Q/K A/B operand loads)
// STORE_T=false -> Y stored [b][M][n] (channel-major, for V tile loads)
// Layout is a template parameter so the store path is branch-free.
// ---------------------------------------------------------------------------
template <int M, bool STORE_T>
__global__ __launch_bounds__(32) void proj_gemm(
    const float* __restrict__ X, const float* __restrict__ W,
    const float* __restrict__ bias, __bf16* __restrict__ Y) {
  const int K = CH, N = NPIX;
  const int lane = threadIdx.x & 31;
  const int l16 = lane & 15;
  const int hs  = lane >> 4;      // half select (0/1)
  const int kl  = hs * 8;
  const int b     = blockIdx.z;
  const int nbase = blockIdx.x * 16;
  const int mbase = blockIdx.y * 16;

  const float* Xb = X + (size_t)b * K * N;
  v8f acc = {};
  for (int k0 = 0; k0 < K; k0 += 32) {
    // A operand: row m = mbase+l16, K elems kl+0..7 and kl+16..23
    v16bf a;
    const float* wrow = W + (size_t)(mbase + l16) * K + (k0 + kl);
#pragma unroll
    for (int e = 0; e < 8; ++e) {
      a[e]     = (__bf16)wrow[e];
      a[8 + e] = (__bf16)wrow[16 + e];
    }
    // B operand: col n = nbase+l16, K elems kl+0..7 and kl+16..23 (stride N)
    v16bf bb;
    const float* xcol = Xb + (size_t)(k0 + kl) * N + (nbase + l16);
#pragma unroll
    for (int e = 0; e < 8; ++e) {
      bb[e]     = (__bf16)xcol[(size_t)e * N];
      bb[8 + e] = (__bf16)xcol[(size_t)(16 + e) * N];
    }
    acc = __builtin_amdgcn_wmma_f32_16x16x32_bf16(false, a, false, bb,
                                                  (short)0, acc, false, false);
  }
  // D layout: VGPR r holds row m = mbase + r + hs*8, col n = nbase + l16
#pragma unroll
  for (int r = 0; r < 8; ++r) {
    const int m = mbase + r + hs * 8;
    const int n = nbase + l16;
    const float val = acc[r] + bias[m];
    if (STORE_T)
      Y[((size_t)b * N + n) * (size_t)M + m] = (__bf16)val;
    else
      Y[((size_t)b * M + m) * (size_t)N + n] = (__bf16)val;
  }
}

// ---------------------------------------------------------------------------
// Fused flash-attention + residual.
// Grid: (NPIX/64, B), block = 128 (4 waves). Wave w owns query rows
// ibase..ibase+15 and a 16x256 f32 accumulator (16 v8f tiles in D layout).
// Per 32-wide j-block: stage K-tile/V-tile to LDS (async), S = Q*K^T (2 WMMA),
// online softmax (half-wave shuffle reductions), O += P*V^T (16 WMMA).
// ---------------------------------------------------------------------------
__global__ __launch_bounds__(128) void attn_kernel(
    const float* __restrict__ x, const __bf16* __restrict__ qb,
    const __bf16* __restrict__ kb, const __bf16* __restrict__ vb,
    float* __restrict__ out) {
  __shared__ __bf16 sK[32 * 32];         // [j_local][hid]  (rows of 64B)
  __shared__ __bf16 sV[CH * 32];         // [c][j_local]    (rows of 64B)
  __shared__ __bf16 sP[4 * 16 * 32];     // per-wave P transpose scratch

  const int tid  = threadIdx.x;
  const int wave = tid >> 5;
  const int lane = tid & 31;
  const int l16  = lane & 15;
  const int hs   = lane >> 4;
  const int kl   = hs * 8;
  const int b    = blockIdx.y;
  const int ibase = blockIdx.x * 64 + wave * 16;

  // Q as WMMA A operand: lane row i = ibase+l16, K(hid) kl+0..7 / kl+16..23
  const __bf16* qrow = qb + ((size_t)b * NPIX + (ibase + l16)) * HID + kl;
  const v16bf aq = cat16(*(const v8bf*)qrow, *(const v8bf*)(qrow + 16));

  v8f acc[16];
#pragma unroll
  for (int t = 0; t < 16; ++t) acc[t] = (v8f){};
  float mst[8], lst[8];
#pragma unroll
  for (int r = 0; r < 8; ++r) { mst[r] = -3.0e38f; lst[r] = 0.0f; }

  __bf16* myP = sP + wave * (16 * 32);

  for (int j0 = 0; j0 < NPIX; j0 += 32) {
    // ---- stage K tile (32x32 bf16 = 2KB) and V tile (256x32 bf16 = 16KB) ----
    {
      const int row = tid >> 2;            // 0..31
      const int o   = (tid & 3) * 8;       // bf16 offset within 64B row
      copy16_to_lds(sK + row * 32 + o,
                    kb + ((size_t)b * NPIX + (j0 + row)) * HID + o);
#pragma unroll
      for (int p = 0; p < 8; ++p) {
        const int t2 = tid + p * 128;      // 0..1023 -> 256 rows x 4 chunks
        const int c  = t2 >> 2;
        const int o2 = (t2 & 3) * 8;
        copy16_to_lds(sV + c * 32 + o2,
                      vb + ((size_t)b * CH + c) * NPIX + j0 + o2);
      }
      async_copies_wait();
    }
    __syncthreads();

    // ---- S = Q * K^T for columns j0..j0+31 (two 16x16 WMMAs) ----
    const __bf16* kr0 = sK + (size_t)l16 * 32 + kl;          // cols j0+0..15
    const __bf16* kr1 = sK + (size_t)(16 + l16) * 32 + kl;   // cols j0+16..31
    const v16bf bk0 = cat16(*(const v8bf*)kr0, *(const v8bf*)(kr0 + 16));
    const v16bf bk1 = cat16(*(const v8bf*)kr1, *(const v8bf*)(kr1 + 16));
    v8f zero = {};
    v8f s0 = __builtin_amdgcn_wmma_f32_16x16x32_bf16(false, aq, false, bk0,
                                                     (short)0, zero, false, false);
    v8f s1 = __builtin_amdgcn_wmma_f32_16x16x32_bf16(false, aq, false, bk1,
                                                     (short)0, zero, false, false);

    // ---- online softmax (row stats live per-lane, uniform per 16-lane half) --
    float scale[8];
#pragma unroll
    for (int r = 0; r < 8; ++r) {
      float tm = fmaxf(s0[r], s1[r]);
#pragma unroll
      for (int off = 8; off >= 1; off >>= 1)
        tm = fmaxf(tm, __shfl_xor(tm, off, 32));
      const float mn = fmaxf(mst[r], tm);
      scale[r] = __expf(mst[r] - mn);
      const float p0 = __expf(s0[r] - mn);
      const float p1 = __expf(s1[r] - mn);
      float ts = p0 + p1;
#pragma unroll
      for (int off = 8; off >= 1; off >>= 1)
        ts += __shfl_xor(ts, off, 32);
      lst[r] = lst[r] * scale[r] + ts;
      mst[r] = mn;
      // write P (D layout) to LDS for the A-layout reload
      const int irow = r + hs * 8;
      myP[irow * 32 + l16]      = (__bf16)p0;
      myP[irow * 32 + 16 + l16] = (__bf16)p1;
    }

    // rescale running accumulators (rows match D layout exactly)
#pragma unroll
    for (int t = 0; t < 16; ++t)
#pragma unroll
      for (int r = 0; r < 8; ++r) acc[t][r] *= scale[r];

    // ---- reload P as WMMA A operand (16x32, K = j_local) ----
    const __bf16* prow = myP + (size_t)l16 * 32 + kl;
    const v16bf ap = cat16(*(const v8bf*)prow, *(const v8bf*)(prow + 16));

    // ---- O += P * V^T over 16 column tiles of C ----
#pragma unroll
    for (int t = 0; t < 16; ++t) {
      const __bf16* vr = sV + (size_t)(t * 16 + l16) * 32 + kl;
      const v16bf bv16 = cat16(*(const v8bf*)vr, *(const v8bf*)(vr + 16));
      acc[t] = __builtin_amdgcn_wmma_f32_16x16x32_bf16(false, ap, false, bv16,
                                                       (short)0, acc[t], false, false);
    }
    __syncthreads();   // before next iteration overwrites sK/sV
  }

  // ---- epilogue: normalize, add residual, store f32 ----
  float inv[8];
#pragma unroll
  for (int r = 0; r < 8; ++r) inv[r] = 1.0f / lst[r];
#pragma unroll
  for (int t = 0; t < 16; ++t) {
#pragma unroll
    for (int r = 0; r < 8; ++r) {
      const int c = t * 16 + l16;
      const int i = ibase + r + hs * 8;
      const size_t idx = ((size_t)b * CH + c) * NPIX + i;
      out[idx] = acc[t][r] * inv[r] + x[idx];
    }
  }
}

// ---------------------------------------------------------------------------
extern "C" void kernel_launch(void* const* d_in, const int* in_sizes, int n_in,
                              void* d_out, int out_size, void* d_ws, size_t ws_size,
                              hipStream_t stream) {
  (void)in_sizes; (void)n_in; (void)out_size; (void)ws_size;
  const float* x  = (const float*)d_in[0];
  const float* wq = (const float*)d_in[1];
  const float* bq = (const float*)d_in[2];
  const float* wk = (const float*)d_in[3];
  const float* bk = (const float*)d_in[4];
  const float* wv = (const float*)d_in[5];
  const float* bv = (const float*)d_in[6];
  float* out = (float*)d_out;

  char* ws = (char*)d_ws;
  __bf16* qb = (__bf16*)(ws);                                  // [B][N][32]  2 MB
  __bf16* kb = (__bf16*)(ws + (size_t)2 * 1024 * 1024);        // [B][N][32]  2 MB
  __bf16* vb = (__bf16*)(ws + (size_t)4 * 1024 * 1024);        // [B][C][N]  16 MB

  dim3 wblk(32);
  proj_gemm<HID, true><<<dim3(NPIX / 16, HID / 16, BATCH), wblk, 0, stream>>>(x, wq, bq, qb);
  proj_gemm<HID, true><<<dim3(NPIX / 16, HID / 16, BATCH), wblk, 0, stream>>>(x, wk, bk, kb);
  proj_gemm<CH, false><<<dim3(NPIX / 16, CH / 16, BATCH), wblk, 0, stream>>>(x, wv, bv, vb);

  attn_kernel<<<dim3(NPIX / 64, BATCH), dim3(128), 0, stream>>>(x, qb, kb, vb, out);
}
